// Encoder_7636451852810
// MI455X (gfx1250) — compile-verified, matrix operand order checked by reading, not run on hardware
//
#include <hip/hip_runtime.h>
#include <hip/hip_bf16.h>

typedef __attribute__((ext_vector_type(16))) _Float16 v16h;
typedef __attribute__((ext_vector_type(8)))  _Float16 v8h;
typedef __attribute__((ext_vector_type(8)))  float    v8f;

#define NN   65536
#define NCON 24576
#define EE   524288
#define SS   256
#define F1   128   // H1 == H2
#define F3   256   // H3
#define F4   128   // H4
#define BN_EPS 1e-5f
#define THRE 0.7f

// ======================= WMMA strip GEMM =======================
// C[M, NT*16](f32) = A[M,K](f16 row-major) @ Bp(packed f16)
// One wave per 16-row strip; NT 16-col tiles accumulated in registers.
// Packed-B layout: fragment for (n-tile tn, k-step ks) = 32 lanes x 16
// contiguous halves; lane = n_local + 16*hi holds
//   halves [0..7]  = B[k0+ 8*hi .. +7][n]
//   halves [8..15] = B[k0+16+8*hi .. +7][n]
template <int NT>
__global__ void wmma_gemm_strip(const _Float16* __restrict__ A,
                                const _Float16* __restrict__ Bp,
                                float* __restrict__ C, int M, int K) {
  const int N = NT * 16;
  int wave = (blockIdx.x * blockDim.x + threadIdx.x) >> 5;
  int lane = threadIdx.x & 31;
  if (wave >= (M >> 4)) return;        // wave-uniform: EXEC all-ones for WMMA
  int m  = lane & 15;
  int hi = lane >> 4;
  const int K32 = K >> 5;

  v8f acc[NT];
#pragma unroll
  for (int t = 0; t < NT; ++t) acc[t] = (v8f){};

  const _Float16* arow = A + (size_t)(wave * 16 + m) * K + 8 * hi;
  for (int ks = 0; ks < K32; ++ks) {
    v8h alo = *(const v8h*)(arow);
    v8h ahi = *(const v8h*)(arow + 16);
    v16h a = __builtin_shufflevector(alo, ahi, 0, 1, 2, 3, 4, 5, 6, 7,
                                     8, 9, 10, 11, 12, 13, 14, 15);
    arow += 32;
#pragma unroll
    for (int tn = 0; tn < NT; ++tn) {
      const v16h b = *(const v16h*)(Bp + (((size_t)tn * K32 + ks) * 32 + lane) * 16);
      acc[tn] = __builtin_amdgcn_wmma_f32_16x16x32_f16(false, a, false, b,
                                                       (short)0, acc[tn], false, false);
    }
  }

  float* cbase = C + (size_t)(wave * 16 + 8 * hi) * N + m;
#pragma unroll
  for (int tn = 0; tn < NT; ++tn) {
    float* crow = cbase + tn * 16;
#pragma unroll
    for (int r = 0; r < 8; ++r) crow[(size_t)r * N] = acc[tn][r];
  }
}

// map (k within 32, n within 16) -> (lane, half) of packed fragment
__device__ __forceinline__ size_t pack_index(int k, int n, int K, int N) {
  int tn = n >> 4, nl = n & 15;
  int ks = k >> 5, kin = k & 31;
  int hi = (kin >> 3) & 1;
  int t  = (kin & 7) + ((kin & 16) >> 1);   // +8 if kin>=16
  int lane = nl + (hi << 4);
  return (((size_t)tn * (K >> 5) + ks) * 32 + lane) * 16 + t;
}

// pack W[O,I](f32) as B with B[k][n] = W[n][k]  (K=I, N=O)
__global__ void pack_wT(const float* __restrict__ W, _Float16* __restrict__ Bp,
                        int O, int I) {
  int idx = blockIdx.x * blockDim.x + threadIdx.x;
  if (idx >= O * I) return;
  int n = idx / I, k = idx % I;
  Bp[pack_index(k, n, I, O)] = (_Float16)W[idx];
}

// pack X[K,N](f32 row-major) as B
__global__ void pack_f32(const float* __restrict__ X, _Float16* __restrict__ Bp,
                         int K, int N) {
  int idx = blockIdx.x * blockDim.x + threadIdx.x;
  if (idx >= K * N) return;
  int k = idx / N, n = idx % N;
  Bp[pack_index(k, n, K, N)] = (_Float16)X[idx];
}

// ======================= misc helpers =======================
__global__ void zero2(float* a, float* b, int n) {
  int i = blockIdx.x * blockDim.x + threadIdx.x;
  if (i < n) { a[i] = 0.0f; b[i] = 0.0f; }
}

// A_hat^T (f16 row-major, used as WMMA A operand) + dinv2 via LDS reduction
__global__ void build_ahat(const float* __restrict__ scen, _Float16* __restrict__ ahT,
                           float* __restrict__ dinv2) {
  __shared__ float red[SS];
  int i = blockIdx.x;       // row of A_hat^T == column i of A_hat
  int k = threadIdx.x;      // row k of A_hat
  float a = (scen[(size_t)k * SS + i] >= THRE) ? 1.0f : 0.0f;
  if (k == i) a += 1.0f;    // self loop
  ahT[(size_t)i * SS + k] = (_Float16)a;
  red[k] = a;
  __syncthreads();
  for (int off = SS / 2; off > 0; off >>= 1) {
    if (k < off) red[k] += red[k + off];
    __syncthreads();
  }
  if (k == 0) {
    float d = red[0];
    dinv2[i] = d > 0.0f ? rsqrtf(d) : 0.0f;
  }
}

__global__ void mlp_kernel(const float* __restrict__ x,
                           const float* __restrict__ vw, const float* __restrict__ vb,
                           const float* __restrict__ cw, const float* __restrict__ cb,
                           float* __restrict__ h) {
  int idx = blockIdx.x * blockDim.x + threadIdx.x;
  if (idx >= NN * F1) return;
  int i = idx >> 7, j = idx & 127;
  float v;
  if (i < NCON) {
    v = x[(size_t)i * 6] * cw[j] + cb[j];
  } else {
    const float* xr = x + (size_t)i * 6;
    v = vb[j];
#pragma unroll
    for (int d = 0; d < 6; ++d) v += xr[d] * vw[j * 6 + d];
  }
  h[idx] = tanhf(v);
}

__global__ void col_stats(const float* __restrict__ X, int rows, int cols,
                          int rows_per_block, float* __restrict__ sum,
                          float* __restrict__ sumsq) {
  int j = threadIdx.x;
  int r0 = blockIdx.x * rows_per_block;
  int r1 = r0 + rows_per_block; if (r1 > rows) r1 = rows;
  float s = 0.0f, s2 = 0.0f;
  for (int i = r0; i < r1; ++i) {
    float v = X[(size_t)i * cols + j];
    s += v; s2 += v * v;
  }
  atomicAdd(&sum[j], s);
  atomicAdd(&sumsq[j], s2);
}

__global__ void bn_finalize(const float* __restrict__ sum, const float* __restrict__ sumsq,
                            const float* __restrict__ g, const float* __restrict__ b,
                            int rows, int cols, float* __restrict__ scale,
                            float* __restrict__ shift) {
  int j = blockIdx.x * blockDim.x + threadIdx.x;
  if (j >= cols) return;
  float mu = sum[j] / (float)rows;
  float var = sumsq[j] / (float)rows - mu * mu;
  float sc = g[j] * rsqrtf(var + BN_EPS);
  scale[j] = sc;
  shift[j] = b[j] - mu * sc;
}

__global__ void bn_apply_f16(const float* __restrict__ X, const float* __restrict__ scale,
                             const float* __restrict__ shift, _Float16* __restrict__ Y,
                             int n, int cols) {
  int idx = blockIdx.x * blockDim.x + threadIdx.x;
  if (idx >= n) return;
  int j = idx % cols;
  Y[idx] = (_Float16)(X[idx] * scale[j] + shift[j]);
}

__global__ void bn_tanh_inplace(float* __restrict__ X, const float* __restrict__ scale,
                                const float* __restrict__ shift, int n, int cols) {
  int idx = blockIdx.x * blockDim.x + threadIdx.x;
  if (idx >= n) return;
  int j = idx % cols;
  X[idx] = tanhf(X[idx] * scale[j] + shift[j]);
}

__global__ void init_deg(float* __restrict__ deg) {
  int i = blockIdx.x * blockDim.x + threadIdx.x;
  if (i < NN) deg[i] = 1.0f;          // self-loop weight
}
__global__ void edge_deg(const int* __restrict__ dst, const float* __restrict__ ew,
                         float* __restrict__ deg) {
  int e = blockIdx.x * blockDim.x + threadIdx.x;
  if (e < EE) atomicAdd(&deg[dst[e]], ew[e]);
}
__global__ void make_dinv(const float* __restrict__ deg, float* __restrict__ dinv) {
  int i = blockIdx.x * blockDim.x + threadIdx.x;
  if (i < NN) dinv[i] = deg[i] > 0.0f ? rsqrtf(deg[i]) : 0.0f;
}

__global__ void selfloop_init(const float* __restrict__ xw, const float* __restrict__ dinv,
                              const float* __restrict__ bias, float* __restrict__ agg) {
  int idx = blockIdx.x * blockDim.x + threadIdx.x;
  if (idx >= NN * F1) return;
  int i = idx >> 7, j = idx & 127;
  float di = dinv[i];
  agg[idx] = bias[j] + di * di * xw[idx];
}

__global__ void edge_scatter(const int* __restrict__ src, const int* __restrict__ dst,
                             const float* __restrict__ ew, const float* __restrict__ dinv,
                             const float* __restrict__ xw, float* __restrict__ agg) {
  int tid = blockIdx.x * blockDim.x + threadIdx.x;   // EE * 32 threads, 4 feats each
  int e = tid >> 5;
  if (e >= EE) return;
  int j = (tid & 31) * 4;
  int s = src[e], d = dst[e];
  float nrm = dinv[s] * ew[e] * dinv[d];
  const float4 v = *(const float4*)(xw + (size_t)s * F1 + j);
  float* o = agg + (size_t)d * F1 + j;
  atomicAdd(o + 0, nrm * v.x);
  atomicAdd(o + 1, nrm * v.y);
  atomicAdd(o + 2, nrm * v.z);
  atomicAdd(o + 3, nrm * v.w);
}

__global__ void pool_scale16(const float* __restrict__ h2, const float* __restrict__ dinv2,
                             _Float16* __restrict__ P16) {
  int g = blockIdx.x, j = threadIdx.x;     // 256 blocks x 128 threads
  float s = 0.0f;
  const float* base = h2 + (size_t)g * 256 * F1 + j;
  for (int i = 0; i < 256; ++i) s += base[(size_t)i * F1];
  P16[(size_t)g * F1 + j] = (_Float16)(dinv2[g] * s * (1.0f / 256.0f));
}

__global__ void rowscale_bias(const float* __restrict__ X, const float* __restrict__ rs,
                              const float* __restrict__ bias, float* __restrict__ Y,
                              int rows, int cols) {
  int idx = blockIdx.x * blockDim.x + threadIdx.x;
  if (idx >= rows * cols) return;
  int i = idx / cols, j = idx % cols;
  Y[idx] = rs[i] * X[idx] + bias[j];
}

// BN3 apply + tanh + fold dinv2 row scale, emitting row-major f16 A operand for next GEMM
__global__ void bn_tanh_scale_pack(const float* __restrict__ X, const float* __restrict__ scale,
                                   const float* __restrict__ shift, const float* __restrict__ rs,
                                   _Float16* __restrict__ Yrow,  // row-major (A operand)
                                   int rows, int cols) {
  int idx = blockIdx.x * blockDim.x + threadIdx.x;
  if (idx >= rows * cols) return;
  int i = idx / cols, j = idx % cols;
  Yrow[idx] = (_Float16)(rs[i] * tanhf(X[idx] * scale[j] + shift[j]));
}

__global__ void final_feat(const float* __restrict__ X, const float* __restrict__ rs,
                           const float* __restrict__ bias, float* __restrict__ out,
                           int rows, int cols) {
  int idx = blockIdx.x * blockDim.x + threadIdx.x;
  if (idx >= rows * cols) return;
  int i = idx / cols, j = idx % cols;
  out[idx] = tanhf(rs[i] * X[idx] + bias[j]);
}

__global__ void col_mean(const float* __restrict__ X, float* __restrict__ out,
                         int rows, int cols) {
  int j = threadIdx.x;
  if (j >= cols) return;
  float s = 0.0f;
  for (int i = 0; i < rows; ++i) s += X[(size_t)i * cols + j];
  out[j] = s / (float)rows;
}

// ======================= host orchestration =======================
extern "C" void kernel_launch(void* const* d_in, const int* in_sizes, int n_in,
                              void* d_out, int out_size, void* d_ws, size_t ws_size,
                              hipStream_t stream) {
  const float* x          = (const float*)d_in[0];
  const float* edge_attr  = (const float*)d_in[1];
  const float* scen_adj   = (const float*)d_in[2];
  const float* mlp_var_w  = (const float*)d_in[3];
  const float* mlp_var_b  = (const float*)d_in[4];
  const float* mlp_con_w  = (const float*)d_in[5];
  const float* mlp_con_b  = (const float*)d_in[6];
  const float* conv1_w    = (const float*)d_in[7];
  const float* conv1_b    = (const float*)d_in[8];
  const float* conv2_w    = (const float*)d_in[9];
  const float* conv2_b    = (const float*)d_in[10];
  const float* conv3_w    = (const float*)d_in[11];
  const float* conv3_b    = (const float*)d_in[12];
  const float* bn1_g      = (const float*)d_in[13];
  const float* bn1_b      = (const float*)d_in[14];
  const float* bn2_g      = (const float*)d_in[15];
  const float* bn2_b      = (const float*)d_in[16];
  const float* bn3_g      = (const float*)d_in[17];
  const float* bn3_b      = (const float*)d_in[18];
  const int*   edge_index = (const int*)d_in[20];
  const int*   e_src = edge_index;
  const int*   e_dst = edge_index + EE;

  char* ws = (char*)d_ws;
  float*    h   = (float*)ws;                           // 32 MB, reused as agg
  _Float16* h16 = (_Float16*)(ws + ((size_t)32 << 20)); // 16 MB
  float*    xw  = (float*)(ws + ((size_t)48 << 20));    // 32 MB
  char* p = ws + ((size_t)80 << 20);
  auto alloc = [&](size_t bytes) { void* r = (void*)p; p += (bytes + 255) & ~(size_t)255; return r; };

  float*    deg   = (float*)alloc(NN * 4);
  float*    dinv  = (float*)alloc(NN * 4);
  float*    sum   = (float*)alloc(SS * 4);
  float*    sumsq = (float*)alloc(SS * 4);
  float*    scale = (float*)alloc(SS * 4);
  float*    shift = (float*)alloc(SS * 4);
  float*    dinv2 = (float*)alloc(SS * 4);
  _Float16* ahT   = (_Float16*)alloc(SS * SS * 2);      // row-major A operand
  _Float16* w1p   = (_Float16*)alloc(F1 * F1 * 2);      // packed B
  _Float16* w2p   = (_Float16*)alloc(F1 * F3 * 2);
  _Float16* w3p   = (_Float16*)alloc(F3 * F4 * 2);
  _Float16* P16   = (_Float16*)alloc(SS * F1 * 2);      // row-major A operand
  float*    M1    = (float*)alloc(SS * F3 * 4);
  _Float16* M1p   = (_Float16*)alloc(SS * F3 * 2);      // packed B
  float*    M2    = (float*)alloc(SS * F3 * 4);
  float*    PRE   = (float*)alloc(SS * F3 * 4);
  _Float16* F16b  = (_Float16*)alloc(SS * F3 * 2);      // row-major A operand
  float*    M3    = (float*)alloc(SS * F4 * 4);
  _Float16* M3p   = (_Float16*)alloc(SS * F4 * 2);      // packed B
  float*    M4    = (float*)alloc(SS * F4 * 4);
  float*    agg   = h;
  float*    out   = (float*)d_out;

  // weight prep + adjacency
  pack_wT<<<(F1 * F1 + 255) / 256, 256, 0, stream>>>(conv1_w, w1p, F1, F1);
  pack_wT<<<(F3 * F1 + 255) / 256, 256, 0, stream>>>(conv2_w, w2p, F3, F1);
  pack_wT<<<(F4 * F3 + 255) / 256, 256, 0, stream>>>(conv3_w, w3p, F4, F3);
  build_ahat<<<SS, SS, 0, stream>>>(scen_adj, ahT, dinv2);

  // MLP + BN1 -> f16
  mlp_kernel<<<(NN * F1) / 256, 256, 0, stream>>>(x, mlp_var_w, mlp_var_b,
                                                  mlp_con_w, mlp_con_b, h);
  zero2<<<1, F1, 0, stream>>>(sum, sumsq, F1);
  col_stats<<<NN / 256, F1, 0, stream>>>(h, NN, F1, 256, sum, sumsq);
  bn_finalize<<<1, F1, 0, stream>>>(sum, sumsq, bn1_g, bn1_b, NN, F1, scale, shift);
  bn_apply_f16<<<(NN * F1) / 256, 256, 0, stream>>>(h, scale, shift, h16, NN * F1, F1);

  // degrees
  init_deg<<<NN / 256, 256, 0, stream>>>(deg);
  edge_deg<<<EE / 256, 256, 0, stream>>>(e_dst, edge_attr, deg);
  make_dinv<<<NN / 256, 256, 0, stream>>>(deg, dinv);

  // xw = h16 @ W1^T  (M=65536, N=128, K=128): 4096 waves, NT=8
  wmma_gemm_strip<8><<<(NN / 16) * 32 / 256, 256, 0, stream>>>(h16, w1p, xw, NN, F1);

  // sparse GCN aggregate
  selfloop_init<<<(NN * F1) / 256, 256, 0, stream>>>(xw, dinv, conv1_b, agg);
  edge_scatter<<<(EE * 32) / 256, 256, 0, stream>>>(e_src, e_dst, edge_attr, dinv, xw, agg);

  // BN2 + tanh
  zero2<<<1, F1, 0, stream>>>(sum, sumsq, F1);
  col_stats<<<NN / 256, F1, 0, stream>>>(agg, NN, F1, 256, sum, sumsq);
  bn_finalize<<<1, F1, 0, stream>>>(sum, sumsq, bn2_g, bn2_b, NN, F1, scale, shift);
  bn_tanh_inplace<<<(NN * F1) / 256, 256, 0, stream>>>(agg, scale, shift, NN * F1, F1);

  // pool (+ fold dinv2 row scale) -> f16 A operand
  pool_scale16<<<SS, F1, 0, stream>>>(agg, dinv2, P16);

  // dense GCN layer 1: M1 = P16 @ W2^T (256x256, K=128); M2 = A_hat^T @ M1 (K=256)
  wmma_gemm_strip<16><<<(SS / 16) * 32 / 256, 256, 0, stream>>>(P16, w2p, M1, SS, F1);
  pack_f32<<<(SS * F3) / 256, 256, 0, stream>>>(M1, M1p, SS, F3);
  wmma_gemm_strip<16><<<(SS / 16) * 32 / 256, 256, 0, stream>>>(ahT, M1p, M2, SS, SS);
  rowscale_bias<<<(SS * F3) / 256, 256, 0, stream>>>(M2, dinv2, conv2_b, PRE, SS, F3);

  // BN3 + tanh (+ fold dinv2 for next layer)
  zero2<<<1, F3, 0, stream>>>(sum, sumsq, F3);
  col_stats<<<1, F3, 0, stream>>>(PRE, SS, F3, SS, sum, sumsq);
  bn_finalize<<<1, F3, 0, stream>>>(sum, sumsq, bn3_g, bn3_b, SS, F3, scale, shift);
  bn_tanh_scale_pack<<<(SS * F3) / 256, 256, 0, stream>>>(PRE, scale, shift, dinv2,
                                                          F16b, SS, F3);

  // dense GCN layer 2: M3 = F16b @ W3^T (256x128, K=256); M4 = A_hat^T @ M3 (K=256)
  wmma_gemm_strip<8><<<(SS / 16) * 32 / 256, 256, 0, stream>>>(F16b, w3p, M3, SS, F3);
  pack_f32<<<(SS * F4) / 256, 256, 0, stream>>>(M3, M3p, SS, F4);
  wmma_gemm_strip<8><<<(SS / 16) * 32 / 256, 256, 0, stream>>>(ahT, M3p, M4, SS, SS);

  // outputs: feat [256,128] then column mean [128]
  final_feat<<<(SS * F4) / 256, 256, 0, stream>>>(M4, dinv2, conv3_b, out, SS, F4);
  col_mean<<<1, F4, 0, stream>>>(out, out + SS * F4, SS, F4);
}
